// CausalSelfAttention_76235669504642
// MI455X (gfx1250) — matrix operand order repaired and checked by reading, not compiled, not run on hardware
//
#include <hip/hip_runtime.h>

// ---------------------------------------------------------------- types
typedef __attribute__((ext_vector_type(16))) __bf16 v16bf;
typedef __attribute__((ext_vector_type(2)))  __bf16 v2bf;
typedef __attribute__((ext_vector_type(8)))  float  v8f;
typedef int vi4 __attribute__((vector_size(16)));   // async builtin's b128 unit

#define GLOBAL_AS __attribute__((address_space(1)))
#define LDS_AS    __attribute__((address_space(3)))

#if defined(__has_builtin)
# if __has_builtin(__builtin_amdgcn_global_load_async_to_lds_b128)
#  define HAVE_ASYNC_LDS 1
# endif
# if __has_builtin(__builtin_amdgcn_cvt_pk_bf16_f32)
#  define HAVE_CVT_PK_BF16 1
# endif
#endif
#ifndef HAVE_ASYNC_LDS
# define HAVE_ASYNC_LDS 0
#endif
#ifndef HAVE_CVT_PK_BF16
# define HAVE_CVT_PK_BF16 0
#endif

union Frag {                 // one WMMA A/B operand (16 bf16 across 8 dwords)
    v16bf        v;
    unsigned int u[8];
};

__device__ __forceinline__ v8f wmma_bf16(const Frag& a, const Frag& b, v8f c) {
    return __builtin_amdgcn_wmma_f32_16x16x32_bf16(false, a.v, false, b.v,
                                                   (short)0, c, false, false);
}

// pack two f32 -> bf16 pair (one v_cvt_pk_bf16_f32 when available)
__device__ __forceinline__ unsigned int pk_bf16(float a, float b) {
#if HAVE_CVT_PK_BF16
    union { v2bf h; unsigned int u; } cv;
    cv.h = __builtin_amdgcn_cvt_pk_bf16_f32(a, b);
    return cv.u;
#else
    // single v_perm_b32 byte-pack (truncation rounding)
    return __builtin_amdgcn_perm(__float_as_uint(b), __float_as_uint(a),
                                 0x07060302u);
#endif
}
__device__ __forceinline__ unsigned short bf16_1(float a) {
#if HAVE_CVT_PK_BF16
    union { v2bf h; unsigned int u; } cv;
    cv.h = __builtin_amdgcn_cvt_pk_bf16_f32(a, a);
    return (unsigned short)(cv.u & 0xFFFFu);
#else
    return (unsigned short)(__float_as_uint(a) >> 16);
#endif
}

__device__ __forceinline__ float redmax16(float v) {
    v = fmaxf(v, __shfl_xor(v, 1, 32));
    v = fmaxf(v, __shfl_xor(v, 2, 32));
    v = fmaxf(v, __shfl_xor(v, 4, 32));
    v = fmaxf(v, __shfl_xor(v, 8, 32));
    return v;
}
__device__ __forceinline__ float redsum16(float v) {
    v += __shfl_xor(v, 1, 32);
    v += __shfl_xor(v, 2, 32);
    v += __shfl_xor(v, 4, 32);
    v += __shfl_xor(v, 8, 32);
    return v;
}

#if HAVE_ASYNC_LDS
__device__ __forceinline__ void wait_async0() {
# if __has_builtin(__builtin_amdgcn_s_wait_asynccnt)
    __builtin_amdgcn_s_wait_asynccnt(0);
# else
    asm volatile("s_wait_asynccnt 0x0" ::: "memory");
# endif
}
__device__ __forceinline__ void async_cp_b128(const void* g, void* l) {
    __builtin_amdgcn_global_load_async_to_lds_b128(
        (GLOBAL_AS vi4*)g, (LDS_AS vi4*)l, 0, 0);
}
#endif

// ---------------------------------------------------------------- sizes
#define BB 2
#define TT 2048
#define CH 1024
#define NH 16
#define HD 64
#define N3 3072

// ================================================================ kernel 1
// qkv = x @ Wqkv + bqkv ; scatter bf16: Q,K -> [B,H,T,D], V -> [B,H,D,T]
// (V pre-transposed so attention can async-copy contiguous key rows)
__global__ __launch_bounds__(256) void qkv_gemm(
    const float* __restrict__ x, const float* __restrict__ Wqkv,
    const float* __restrict__ bqkv,
    unsigned short* __restrict__ Qb, unsigned short* __restrict__ Kb,
    unsigned short* __restrict__ Vb)
{
    __shared__ unsigned int lds_a[128 * 20];   // 128 rows x 32 K (bf16 pairs)
    __shared__ unsigned int lds_b[64 * 20];    // 64 cols  x 32 K (bf16 pairs)

    const int tid  = threadIdx.x;
    const int lane = tid & 31, wave = tid >> 5;
    const int wm = wave >> 1, wn = wave & 1;     // 4x2 waves -> 128x64 tile
    const int l15 = lane & 15, half = lane >> 4;
    const int m0 = blockIdx.x * 128, n0 = blockIdx.y * 64;

    // marching base pointers (per-thread), immediate offsets inside the loop
    const float* ap = x + (long)(m0 + (tid >> 4)) * CH + ((tid & 15) << 1);
    const float* bp = Wqkv + (long)((tid >> 6) << 1) * N3 + n0 + (tid & 63);
    const unsigned la = (unsigned)((tid >> 4) * 20 + (tid & 15));
    const unsigned lb = (unsigned)((tid & 63) * 20 + (tid >> 6));

    v8f z = {0.f, 0.f, 0.f, 0.f, 0.f, 0.f, 0.f, 0.f};
    v8f acc[2][2] = {{z, z}, {z, z}};

    for (int k0 = 0; k0 < CH; k0 += 32) {
#pragma unroll
        for (int i = 0; i < 8; ++i) {          // A tile 128x32
            float2 f = *(const float2*)(ap + (long)i * 16 * CH);
            lds_a[la + i * 320] = pk_bf16(f.x, f.y);
        }
#pragma unroll
        for (int i = 0; i < 4; ++i) {          // B tile 32x64 (pairs along K)
            float w0 = bp[(long)i * 8 * N3];
            float w1 = bp[(long)i * 8 * N3 + N3];
            lds_b[lb + i * 4] = pk_bf16(w0, w1);
        }
        if (k0 + 32 < CH) __builtin_prefetch(ap + 32, 0, 1);
        ap += 32;
        bp += (long)32 * N3;
        __syncthreads();

        Frag af[2], bfr[2];
#pragma unroll
        for (int i = 0; i < 2; ++i) {
            int m = wm * 32 + i * 16 + l15;
            int o = half ? 4 : 0;              // A 16-bit layout (ISA 7.12.2)
#pragma unroll
            for (int u = 0; u < 4; ++u) {
                af[i].u[u]     = lds_a[m * 20 + o + u];
                af[i].u[4 + u] = lds_a[m * 20 + o + 8 + u];
            }
        }
#pragma unroll
        for (int j = 0; j < 2; ++j) {
            int n = wn * 32 + j * 16 + l15;
            int o = half ? 4 : 0;              // B: column per lane, K packed
#pragma unroll
            for (int u = 0; u < 4; ++u) {
                bfr[j].u[u]     = lds_b[n * 20 + o + u];
                bfr[j].u[4 + u] = lds_b[n * 20 + o + 8 + u];
            }
        }
#pragma unroll
        for (int i = 0; i < 2; ++i)
#pragma unroll
            for (int j = 0; j < 2; ++j)
                acc[i][j] = wmma_bf16(af[i], bfr[j], acc[i][j]);
        __syncthreads();
    }

    const float scale = 0.125f;  // 1/sqrt(64)
#pragma unroll
    for (int i = 0; i < 2; ++i)
#pragma unroll
        for (int j = 0; j < 2; ++j) {
            int gn = n0 + wn * 32 + j * 16 + l15;
            float bias = bqkv[gn];
            int which = gn >> 10, c = gn & 1023;
            int h = c >> 6, d = c & 63;
            unsigned short* dst = which == 0 ? Qb : (which == 1 ? Kb : Vb);
#pragma unroll
            for (int r = 0; r < 8; ++r) {
                int gm = m0 + wm * 32 + i * 16 + r + (half << 3);
                int b = gm >> 11, t = gm & (TT - 1);
                float v = acc[i][j][r] + bias;
                if (which == 0) v *= scale;
                long idx = (which == 2)
                    ? ((long)(b * NH + h) * HD + d) * TT + t   // V: [B,H,D,T]
                    : ((long)(b * NH + h) * TT + t) * HD + d;  // Q,K: [B,H,T,D]
                dst[idx] = bf16_1(v);
            }
        }
}

// ================================================================ kernel 2
// flash attention: one block = 64 queries of one (b,h); 4 waves x 16 rows
__global__ __launch_bounds__(128) void attn_fwd(
    const unsigned short* __restrict__ Qb, const unsigned short* __restrict__ Kb,
    const unsigned short* __restrict__ Vb, float* __restrict__ attn_out)
{
    __shared__ unsigned int lds_k[64 * 36];       // [key][d-pairs]
    __shared__ unsigned int lds_v[64 * 36];       // [d][key-pairs] (V^T rows)
    __shared__ unsigned int lds_p[4][16 * 36];    // per-wave P tile

    const int tid  = threadIdx.x;
    const int lane = tid & 31, wave = tid >> 5;
    const int l15 = lane & 15, half = lane >> 4;
    const int bh = blockIdx.x;                 // B*H = 32
    const int qb = blockIdx.y;                 // T/64 = 32
    const unsigned short* Qh = Qb + (long)bh * TT * HD;
    const unsigned short* Kh = Kb + (long)bh * TT * HD;
    const unsigned short* Vh = Vb + (long)bh * HD * TT;   // [D,T]
    const int q0 = qb * 64 + wave * 16;

    // ---- Q fragments for this wave's 16 rows (2 chunks over D=64)
    Frag qf[2];
    {
        const unsigned int* qrow = (const unsigned int*)(Qh + (long)(q0 + l15) * HD);
        int o = half ? 4 : 0;
#pragma unroll
        for (int c = 0; c < 2; ++c)
#pragma unroll
            for (int u = 0; u < 4; ++u) {
                qf[c].u[u]     = qrow[c * 16 + o + u];
                qf[c].u[4 + u] = qrow[c * 16 + o + 8 + u];
            }
    }

    // per-thread staging pointers: 16B granules, marched by key-block
    const unsigned short* kgp = Kh + (long)(tid >> 3) * HD + (tid & 7) * 8;
    const unsigned short* vgp = Vh + (long)(tid >> 3) * TT + (tid & 7) * 8;
    unsigned int* klp = &lds_k[(tid >> 3) * 36 + (tid & 7) * 4];
    unsigned int* vlp = &lds_v[(tid >> 3) * 36 + (tid & 7) * 4];

    float mrow[8], lrow[8];
    v8f z = {0.f, 0.f, 0.f, 0.f, 0.f, 0.f, 0.f, 0.f};
    v8f oacc[4] = {z, z, z, z};
#pragma unroll
    for (int r = 0; r < 8; ++r) { mrow[r] = -1e30f; lrow[r] = 0.f; }

    for (int kb = 0; kb <= qb; ++kb) {
        // ---- stage K (natural) and V^T (natural rows) tiles, 16B per lane op
#if HAVE_ASYNC_LDS
#pragma unroll
        for (int i = 0; i < 4; ++i) {
            async_cp_b128(kgp + (long)i * 16 * HD, klp + i * 16 * 36);
            async_cp_b128(vgp + (long)i * 16 * TT, vlp + i * 16 * 36);
        }
        wait_async0();
#else
#pragma unroll
        for (int i = 0; i < 4; ++i) {
            uint4 kv = *(const uint4*)(kgp + (long)i * 16 * HD);
            uint4 vv = *(const uint4*)(vgp + (long)i * 16 * TT);
            unsigned int* kr = klp + i * 16 * 36;
            unsigned int* vr = vlp + i * 16 * 36;
            kr[0] = kv.x; kr[1] = kv.y; kr[2] = kv.z; kr[3] = kv.w;
            vr[0] = vv.x; vr[1] = vv.y; vr[2] = vv.z; vr[3] = vv.w;
        }
#endif
        kgp += 64 * HD;   // next 64 keys
        vgp += 64;
        __syncthreads();

        // ---- S = Q @ K^T for 4 key sub-tiles
        v8f s[4];
#pragma unroll
        for (int a = 0; a < 4; ++a) {
            s[a] = z;
#pragma unroll
            for (int c = 0; c < 2; ++c) {
                Frag kf;
                int key = a * 16 + l15;
                int o = half ? 4 : 0;          // d-pairs within 32-chunk
#pragma unroll
                for (int u = 0; u < 4; ++u) {
                    kf.u[u]     = lds_k[key * 36 + c * 16 + o + u];
                    kf.u[4 + u] = lds_k[key * 36 + c * 16 + o + 8 + u];
                }
                s[a] = wmma_bf16(qf[c], kf, s[a]);
            }
        }
        // ---- causal mask (diagonal block only)
        if (kb == qb) {
#pragma unroll
            for (int a = 0; a < 4; ++a)
#pragma unroll
                for (int r = 0; r < 8; ++r) {
                    int key_l = a * 16 + l15;
                    int q_l = wave * 16 + r + (half << 3);
                    if (key_l > q_l) s[a][r] = -1e30f;
                }
        }
        // ---- online softmax update
        float csc[8], rsum[8];
#pragma unroll
        for (int r = 0; r < 8; ++r) {
            float rm = fmaxf(fmaxf(s[0][r], s[1][r]), fmaxf(s[2][r], s[3][r]));
            rm = redmax16(rm);
            float mn = fmaxf(mrow[r], rm);
            csc[r] = __expf(mrow[r] - mn);
            mrow[r] = mn;
            rsum[r] = 0.f;
        }
#pragma unroll
        for (int a = 0; a < 4; ++a)
#pragma unroll
            for (int r = 0; r < 8; ++r) {
                float p = __expf(s[a][r] - mrow[r]);
                s[a][r] = p;
                rsum[r] += p;
            }
#pragma unroll
        for (int r = 0; r < 8; ++r)
            lrow[r] = lrow[r] * csc[r] + redsum16(rsum[r]);

        // ---- P -> LDS (C-layout -> A-layout), pair-packed via lane shuffle
#pragma unroll
        for (int a = 0; a < 4; ++a)
#pragma unroll
            for (int r = 0; r < 8; ++r) {
                float p = s[a][r];
                float po = __shfl_xor(p, 1, 32);
                if ((l15 & 1) == 0)
                    lds_p[wave][(r + (half << 3)) * 36 + a * 8 + (l15 >> 1)] =
                        pk_bf16(p, po);
            }
        // ---- rescale O, then O += P @ V
#pragma unroll
        for (int a = 0; a < 4; ++a)
#pragma unroll
            for (int r = 0; r < 8; ++r) oacc[a][r] *= csc[r];
#pragma unroll
        for (int c = 0; c < 2; ++c) {          // key chunks of 32
            Frag pf;
            int o = half ? 4 : 0;
#pragma unroll
            for (int u = 0; u < 4; ++u) {
                pf.u[u]     = lds_p[wave][l15 * 36 + c * 16 + o + u];
                pf.u[4 + u] = lds_p[wave][l15 * 36 + c * 16 + o + 8 + u];
            }
#pragma unroll
            for (int a = 0; a < 4; ++a) {
                Frag vf;
                int ov = half ? 4 : 0;
#pragma unroll
                for (int u = 0; u < 4; ++u) {
                    vf.u[u]     = lds_v[(a * 16 + l15) * 36 + c * 16 + ov + u];
                    vf.u[4 + u] = lds_v[(a * 16 + l15) * 36 + c * 16 + ov + 8 + u];
                }
                oacc[a] = wmma_bf16(pf, vf, oacc[a]);
            }
        }
        __syncthreads();
    }

    // ---- normalize and write [B,T,C] f32
    const int b = bh >> 4, h = bh & 15;
#pragma unroll
    for (int r = 0; r < 8; ++r) {
        float inv = 1.0f / lrow[r];
        int q = q0 + r + (half << 3);
#pragma unroll
        for (int a = 0; a < 4; ++a) {
            int d = a * 16 + l15;
            attn_out[((long)(b * TT + q)) * CH + h * HD + d] = oacc[a][r] * inv;
        }
    }
}

// ================================================================ kernel 3
// out = attn @ Wout + bout  (f32 in, f32 out)
__global__ __launch_bounds__(256) void out_gemm(
    const float* __restrict__ A, const float* __restrict__ Wout,
    const float* __restrict__ bout, float* __restrict__ out)
{
    __shared__ unsigned int lds_a[128 * 20];
    __shared__ unsigned int lds_b[64 * 20];

    const int tid  = threadIdx.x;
    const int lane = tid & 31, wave = tid >> 5;
    const int wm = wave >> 1, wn = wave & 1;
    const int l15 = lane & 15, half = lane >> 4;
    const int m0 = blockIdx.x * 128, n0 = blockIdx.y * 64;

    const float* ap = A + (long)(m0 + (tid >> 4)) * CH + ((tid & 15) << 1);
    const float* bp = Wout + (long)((tid >> 6) << 1) * CH + n0 + (tid & 63);
    const unsigned la = (unsigned)((tid >> 4) * 20 + (tid & 15));
    const unsigned lb = (unsigned)((tid & 63) * 20 + (tid >> 6));

    v8f z = {0.f, 0.f, 0.f, 0.f, 0.f, 0.f, 0.f, 0.f};
    v8f acc[2][2] = {{z, z}, {z, z}};

    for (int k0 = 0; k0 < CH; k0 += 32) {
#pragma unroll
        for (int i = 0; i < 8; ++i) {
            float2 f = *(const float2*)(ap + (long)i * 16 * CH);
            lds_a[la + i * 320] = pk_bf16(f.x, f.y);
        }
#pragma unroll
        for (int i = 0; i < 4; ++i) {
            float w0 = bp[(long)i * 8 * CH];
            float w1 = bp[(long)i * 8 * CH + CH];
            lds_b[lb + i * 4] = pk_bf16(w0, w1);
        }
        if (k0 + 32 < CH) __builtin_prefetch(ap + 32, 0, 1);
        ap += 32;
        bp += (long)32 * CH;
        __syncthreads();

        Frag af[2], bfr[2];
#pragma unroll
        for (int i = 0; i < 2; ++i) {
            int m = wm * 32 + i * 16 + l15;
            int o = half ? 4 : 0;
#pragma unroll
            for (int u = 0; u < 4; ++u) {
                af[i].u[u]     = lds_a[m * 20 + o + u];
                af[i].u[4 + u] = lds_a[m * 20 + o + 8 + u];
            }
        }
#pragma unroll
        for (int j = 0; j < 2; ++j) {
            int n = wn * 32 + j * 16 + l15;
            int o = half ? 4 : 0;
#pragma unroll
            for (int u = 0; u < 4; ++u) {
                bfr[j].u[u]     = lds_b[n * 20 + o + u];
                bfr[j].u[4 + u] = lds_b[n * 20 + o + 8 + u];
            }
        }
#pragma unroll
        for (int i = 0; i < 2; ++i)
#pragma unroll
            for (int j = 0; j < 2; ++j)
                acc[i][j] = wmma_bf16(af[i], bfr[j], acc[i][j]);
        __syncthreads();
    }

#pragma unroll
    for (int i = 0; i < 2; ++i)
#pragma unroll
        for (int j = 0; j < 2; ++j) {
            int gn = n0 + wn * 32 + j * 16 + l15;
            float bias = bout[gn];
#pragma unroll
            for (int r = 0; r < 8; ++r) {
                int gm = m0 + wm * 32 + i * 16 + r + (half << 3);
                out[(long)gm * CH + gn] = acc[i][j][r] + bias;
            }
        }
}

// ================================================================ launch
extern "C" void kernel_launch(void* const* d_in, const int* in_sizes, int n_in,
                              void* d_out, int out_size, void* d_ws, size_t ws_size,
                              hipStream_t stream) {
    (void)in_sizes; (void)n_in; (void)out_size; (void)ws_size;
    const float* x    = (const float*)d_in[0];
    const float* Wqkv = (const float*)d_in[1];
    const float* bqkv = (const float*)d_in[2];
    const float* Wout = (const float*)d_in[3];
    const float* bout = (const float*)d_in[4];
    float* out = (float*)d_out;

    char* ws = (char*)d_ws;
    unsigned short* Qb = (unsigned short*)(ws);                 //  8 MB
    unsigned short* Kb = (unsigned short*)(ws + 8388608);       //  8 MB
    unsigned short* Vb = (unsigned short*)(ws + 16777216);      //  8 MB (V^T)
    float*          At = (float*)(ws + 25165824);               // 16 MB

    qkv_gemm<<<dim3(32, 48), 256, 0, stream>>>(x, Wqkv, bqkv, Qb, Kb, Vb);
    attn_fwd<<<dim3(32, 32), 128, 0, stream>>>(Qb, Kb, Vb, At);
    out_gemm<<<dim3(32, 16), 256, 0, stream>>>(At, Wout, bout, out);
}
